// GCN_7576322310411
// MI455X (gfx1250) — matrix operand order repaired and checked by reading, not compiled, not run on hardware
//
#include <hip/hip_runtime.h>
#include <hip/hip_bf16.h>

// ---------------- problem constants (match reference) ----------------
#define F_IN   128
#define HEADS  4
#define HID    64
#define C1     (HEADS * HID)    // 256
#define NCLS   16
#define C2     (HEADS * NCLS)   // 64
#define NEG_SLOPE 0.2f

typedef _Float16 h8   __attribute__((ext_vector_type(8)));
typedef _Float16 v16h __attribute__((ext_vector_type(16)));
typedef float    v8f  __attribute__((ext_vector_type(8)));

// ---------------- small utility kernels ----------------
__global__ void cvt_f16_kernel(const float* __restrict__ x, _Float16* __restrict__ y, size_t n) {
  size_t i = (size_t)blockIdx.x * blockDim.x + threadIdx.x;
  if (i < n) y[i] = (_Float16)x[i];
}

// Bt[col][k] = (f16) W[k][col]   (W is [K, Ncols] row-major)
__global__ void transpose_to_f16_kernel(const float* __restrict__ W, _Float16* __restrict__ Bt,
                                        int K, int Ncols) {
  int i = blockIdx.x * blockDim.x + threadIdx.x;
  if (i < K * Ncols) {
    int k = i / Ncols, c = i - k * Ncols;
    Bt[(size_t)c * K + k] = (_Float16)W[i];
  }
}

__global__ void fill_f32_kernel(float* __restrict__ p, float v, size_t n) {
  size_t i = (size_t)blockIdx.x * blockDim.x + threadIdx.x;
  if (i < n) p[i] = v;
}

// out[n*C + c] = bias[c]  (accumulator init; bias added before atomic aggregation)
__global__ void init_bias_kernel(float* __restrict__ out, const float* __restrict__ bias,
                                 size_t n, int C) {
  size_t i = (size_t)blockIdx.x * blockDim.x + threadIdx.x;
  if (i < n) out[i] = bias[i % C];
}

// y = f16(elu(x))
__global__ void elu_to_f16_kernel(const float* __restrict__ x, _Float16* __restrict__ y, size_t n) {
  size_t i = (size_t)blockIdx.x * blockDim.x + threadIdx.x;
  if (i < n) {
    float v = x[i];
    v = (v > 0.0f) ? v : expm1f(v);
    y[i] = (_Float16)v;
  }
}

// ---------------- WMMA GEMM:  C[M,Ncols] = A[M,K](f16) * Bt[Ncols,K](f16) ----------------
// blockDim = 128 (4 wave32).  grid = (Ncols/64, ceil(M/16)).  One 16x16 tile per wave.
// Fragment layouts per CDNA5 ISA 7.12.2 (16-bit A 16x32 / B 32x16 / f32 C 16x16).
__global__ __launch_bounds__(128)
void gemm_wmma_f16_kernel(const _Float16* __restrict__ A, const _Float16* __restrict__ Bt,
                          float* __restrict__ C, int M, int K, int Ncols) {
  const int wave    = threadIdx.x >> 5;
  const int lane    = threadIdx.x & 31;
  const int halfsel = lane >> 4;    // 0: lanes 0-15, 1: lanes 16-31
  const int l15     = lane & 15;
  const int row0    = blockIdx.y * 16;
  const int col0    = blockIdx.x * 64 + wave * 16;

  int arow_i = row0 + l15;
  if (arow_i >= M) arow_i = M - 1;                    // clamp (stores guarded below)
  const _Float16* __restrict__ arow = A  + (size_t)arow_i        * K;
  const _Float16* __restrict__ bcol = Bt + (size_t)(col0 + l15)  * K;

  v8f acc = {};
  for (int kk = 0; kk < K; kk += 32) {
    // A row fragment: halves 0..7 = K[kk + 8*halfsel .. +7], 8..15 = K[kk+16+8*halfsel .. +7]
    h8 alo = *(const h8*)(arow + kk + halfsel * 8);
    h8 ahi = *(const h8*)(arow + kk + halfsel * 8 + 16);
    // B col fragment: halves 0..7 = K[kk + 8*halfsel .. +7], 8..15 = K[kk+16+8*halfsel .. +7] of column (lane&15)
    h8 blo = *(const h8*)(bcol + kk + halfsel * 8);
    h8 bhi = *(const h8*)(bcol + kk + halfsel * 8 + 16);
    v16h a = __builtin_shufflevector(alo, ahi, 0,1,2,3,4,5,6,7,8,9,10,11,12,13,14,15);
    v16h b = __builtin_shufflevector(blo, bhi, 0,1,2,3,4,5,6,7,8,9,10,11,12,13,14,15);
    acc = __builtin_amdgcn_wmma_f32_16x16x32_f16(false, a, false, b, (short)0, acc, false, false);
  }

  // C layout: VGPR v -> row = row0 + v + 8*halfsel, col = col0 + (lane&15)
  #pragma unroll
  for (int v = 0; v < 8; ++v) {
    int r = row0 + halfsel * 8 + v;
    if (r < M) C[(size_t)r * Ncols + col0 + l15] = acc[v];
  }
}

// ---------------- attention dot products ----------------
// asrc[n,h] = sum_c H[n, h*Cper + c] * att_s[h, c]   (and same for adst)
__global__ void att_dots_kernel(const float* __restrict__ H,
                                const float* __restrict__ att_s, const float* __restrict__ att_d,
                                float* __restrict__ asrc, float* __restrict__ adst,
                                int N, int Cper) {
  int t = blockIdx.x * blockDim.x + threadIdx.x;
  if (t >= N * HEADS) return;
  int n = t >> 2, h = t & 3;
  const float* hp = H + (size_t)n * (HEADS * Cper) + h * Cper;
  const float* as = att_s + h * Cper;
  const float* ad = att_d + h * Cper;
  float s = 0.0f, d = 0.0f;
  for (int c = 0; c < Cper; ++c) { float v = hp[c]; s += v * as[c]; d += v * ad[c]; }
  asrc[t] = s; adst[t] = d;
}

// ---------------- edge phase ----------------
__device__ __forceinline__ void edge_sd(const long long* __restrict__ ei, int E, int idx,
                                        int& s, int& d) {
  if (idx < E) { s = (int)ei[idx]; d = (int)ei[(size_t)E + idx]; }
  else         { s = d = idx - E; }          // synthesized self-loop
}

__device__ __forceinline__ void atomicMaxF(float* addr, float val) {
  if (val >= 0.0f) atomicMax((int*)addr, __float_as_int(val));
  else             atomicMin((unsigned int*)addr, (unsigned int)__float_as_int(val));
}

// pass 1: e = leaky_relu(asrc[src]+adst[dst]); store; segment max into mbuf[dst]
__global__ void edge_lrelu_max_kernel(const long long* __restrict__ ei, int E, int ET,
                                      const float* __restrict__ asrc, const float* __restrict__ adst,
                                      float* __restrict__ ebuf, float* __restrict__ mbuf) {
  int idx = blockIdx.x * blockDim.x + threadIdx.x;
  if (idx >= ET) return;
  int s, d; edge_sd(ei, E, idx, s, d);
  #pragma unroll
  for (int h = 0; h < HEADS; ++h) {
    float e = asrc[s * HEADS + h] + adst[d * HEADS + h];
    e = (e > 0.0f) ? e : NEG_SLOPE * e;
    ebuf[(size_t)idx * HEADS + h] = e;
    atomicMaxF(&mbuf[d * HEADS + h], e);
  }
}

// pass 2: ex = exp(e - m[dst]); store; segment sum into dbuf[dst]
__global__ void edge_exp_sum_kernel(const long long* __restrict__ ei, int E, int ET,
                                    float* __restrict__ ebuf, const float* __restrict__ mbuf,
                                    float* __restrict__ dbuf) {
  int idx = blockIdx.x * blockDim.x + threadIdx.x;
  if (idx >= ET) return;
  int s, d; edge_sd(ei, E, idx, s, d);
  (void)s;
  #pragma unroll
  for (int h = 0; h < HEADS; ++h) {
    float ex = __expf(ebuf[(size_t)idx * HEADS + h] - mbuf[d * HEADS + h]);
    ebuf[(size_t)idx * HEADS + h] = ex;
    atomicAdd(&dbuf[d * HEADS + h], ex);
  }
}

// pass 3: out[dst] += (ex/denom[dst]) * H[src]   — one wave32 per edge, coalesced channels
template <int CT>
__global__ __launch_bounds__(256)
void edge_aggregate_kernel(const long long* __restrict__ ei, int E, int ET,
                           const float* __restrict__ ebuf, const float* __restrict__ dbuf,
                           const float* __restrict__ H, float* __restrict__ out) {
  int w = blockIdx.x * (blockDim.x >> 5) + (threadIdx.x >> 5);
  if (w >= ET) return;
  int lane = threadIdx.x & 31;
  int s, d; edge_sd(ei, E, w, s, d);
  #pragma unroll
  for (int ch = lane; ch < CT; ch += 32) {
    int h = ch / (CT / HEADS);
    float alpha = ebuf[(size_t)w * HEADS + h] / dbuf[d * HEADS + h];
    atomicAdd(&out[(size_t)d * CT + ch], H[(size_t)s * CT + ch] * alpha);
  }
}

// ---------------- host-side orchestration ----------------
extern "C" void kernel_launch(void* const* d_in, const int* in_sizes, int n_in,
                              void* d_out, int out_size, void* d_ws, size_t ws_size,
                              hipStream_t stream) {
  const float*     x        = (const float*)d_in[0];
  const long long* ei       = (const long long*)d_in[1];   // int64 edge_index [2,E]
  const float*     W1       = (const float*)d_in[2];
  const float*     att_src1 = (const float*)d_in[3];
  const float*     att_dst1 = (const float*)d_in[4];
  const float*     bias1    = (const float*)d_in[5];
  const float*     W2       = (const float*)d_in[6];
  const float*     att_src2 = (const float*)d_in[7];
  const float*     att_dst2 = (const float*)d_in[8];
  const float*     bias2    = (const float*)d_in[9];
  float*           out      = (float*)d_out;

  const int N  = in_sizes[0] / F_IN;   // 50000
  const int E  = in_sizes[1] / 2;      // 800000
  const int ET = E + N;                // + self loops

  // workspace layout (256B aligned); H1h aliases H1, H2 aliases Xh (byte-exact)
  char* ws = (char*)d_ws; size_t off = 0;
  auto take = [&](size_t bytes) -> char* {
    char* p = ws + off; off = (off + bytes + 255) & ~(size_t)255; return p;
  };
  _Float16* Xh   = (_Float16*)take((size_t)N * F_IN * 2);
  _Float16* W1t  = (_Float16*)take((size_t)C1 * F_IN * 2);
  _Float16* W2t  = (_Float16*)take((size_t)C2 * C1 * 2);
  float*    H1   = (float*)take((size_t)N * C1 * 4);
  float*    agg1 = (float*)take((size_t)N * C1 * 4);
  float*    asrc = (float*)take((size_t)N * HEADS * 4);
  float*    adst = (float*)take((size_t)N * HEADS * 4);
  float*    mbuf = (float*)take((size_t)N * HEADS * 4);
  float*    dbuf = (float*)take((size_t)N * HEADS * 4);
  float*    ebuf = (float*)take((size_t)ET * HEADS * 4);
  _Float16* H1h  = (_Float16*)H1;   // H1 (f32) dead after layer-1 aggregation
  float*    H2   = (float*)Xh;      // Xh dead after GEMM1; N*128*2 == N*64*4 bytes
  (void)n_in; (void)ws_size; (void)out_size;

  const int TB = 256;
  auto nb = [](long long n, int tb) { return (unsigned)((n + tb - 1) / tb); };

  // ---- precision prep ----
  cvt_f16_kernel<<<nb((long long)N * F_IN, TB), TB, 0, stream>>>(x, Xh, (size_t)N * F_IN);
  transpose_to_f16_kernel<<<nb((long long)F_IN * C1, TB), TB, 0, stream>>>(W1, W1t, F_IN, C1);
  transpose_to_f16_kernel<<<nb((long long)C1 * C2, TB), TB, 0, stream>>>(W2, W2t, C1, C2);

  // ---- layer 1 ----
  gemm_wmma_f16_kernel<<<dim3(C1 / 64, (N + 15) / 16), 128, 0, stream>>>(Xh, W1t, H1, N, F_IN, C1);
  att_dots_kernel<<<nb((long long)N * HEADS, TB), TB, 0, stream>>>(H1, att_src1, att_dst1,
                                                                   asrc, adst, N, HID);
  fill_f32_kernel<<<nb((long long)N * HEADS, TB), TB, 0, stream>>>(mbuf, -3.0e38f, (size_t)N * HEADS);
  fill_f32_kernel<<<nb((long long)N * HEADS, TB), TB, 0, stream>>>(dbuf, 0.0f, (size_t)N * HEADS);
  init_bias_kernel<<<nb((long long)N * C1, TB), TB, 0, stream>>>(agg1, bias1, (size_t)N * C1, C1);

  edge_lrelu_max_kernel<<<nb(ET, TB), TB, 0, stream>>>(ei, E, ET, asrc, adst, ebuf, mbuf);
  edge_exp_sum_kernel<<<nb(ET, TB), TB, 0, stream>>>(ei, E, ET, ebuf, mbuf, dbuf);
  edge_aggregate_kernel<C1><<<nb((long long)ET * 32, 256), 256, 0, stream>>>(ei, E, ET, ebuf, dbuf,
                                                                             H1, agg1);
  elu_to_f16_kernel<<<nb((long long)N * C1, TB), TB, 0, stream>>>(agg1, H1h, (size_t)N * C1);

  // ---- layer 2 ----
  gemm_wmma_f16_kernel<<<dim3(C2 / 64, (N + 15) / 16), 128, 0, stream>>>(H1h, W2t, H2, N, C1, C2);
  att_dots_kernel<<<nb((long long)N * HEADS, TB), TB, 0, stream>>>(H2, att_src2, att_dst2,
                                                                   asrc, adst, N, NCLS);
  fill_f32_kernel<<<nb((long long)N * HEADS, TB), TB, 0, stream>>>(mbuf, -3.0e38f, (size_t)N * HEADS);
  fill_f32_kernel<<<nb((long long)N * HEADS, TB), TB, 0, stream>>>(dbuf, 0.0f, (size_t)N * HEADS);
  init_bias_kernel<<<nb((long long)N * C2, TB), TB, 0, stream>>>(out, bias2, (size_t)N * C2, C2);

  edge_lrelu_max_kernel<<<nb(ET, TB), TB, 0, stream>>>(ei, E, ET, asrc, adst, ebuf, mbuf);
  edge_exp_sum_kernel<<<nb(ET, TB), TB, 0, stream>>>(ei, E, ET, ebuf, mbuf, dbuf);
  edge_aggregate_kernel<C2><<<nb((long long)ET * 32, 256), 256, 0, stream>>>(ei, E, ET, ebuf, dbuf,
                                                                             H2, out);
}
